// Mamba_20392504721965
// MI455X (gfx1250) — matrix operand order repaired and checked by reading, not compiled, not run on hardware
//
#include <hip/hip_runtime.h>
#include <hip/hip_bf16.h>

// ---- shapes (fixed by the reference) ----
#define BB 2
#define LL 2048
#define DM 512
#define DI 1024
#define DS 16
#define DTR 32
#define ROWS (BB * LL)          // 4096

typedef __bf16 v16bf __attribute__((ext_vector_type(16)));
typedef __bf16 v8bf  __attribute__((ext_vector_type(8)));
typedef float  v8f   __attribute__((ext_vector_type(8)));
typedef int    v4i   __attribute__((ext_vector_type(4)));

// gfx1250 async-load-to-LDS path (guarded: falls back to plain LDS stores)
#if defined(__gfx1250__) && __has_builtin(__builtin_amdgcn_global_load_async_to_lds_b128) && \
    __has_builtin(__builtin_amdgcn_s_wait_asynccnt)
#define HAVE_ASYNC_LDS 1
typedef __attribute__((address_space(1))) v4i* glob_v4i_p;
typedef __attribute__((address_space(3))) v4i* lds_v4i_p;
#endif

// ------------------------------------------------------------------
// generic f32 -> bf16 cast
// ------------------------------------------------------------------
__global__ void cast_f32_bf16(const float* __restrict__ in, __bf16* __restrict__ out, int n) {
    int i = blockIdx.x * blockDim.x + threadIdx.x;
    if (i < n) out[i] = (__bf16)in[i];
}

// ------------------------------------------------------------------
// RMSNorm: one block per row of x (ROWS x DM), emits bf16 normalized row
// ------------------------------------------------------------------
__global__ void rmsnorm_kernel(const float* __restrict__ x, const float* __restrict__ w,
                               __bf16* __restrict__ xn) {
    __shared__ float red[256];
    const int row = blockIdx.x, tid = threadIdx.x;
    const float* xr = x + (size_t)row * DM;
    float s = 0.f;
    for (int c = tid; c < DM; c += 256) { float v = xr[c]; s += v * v; }
    red[tid] = s; __syncthreads();
    for (int off = 128; off > 0; off >>= 1) {
        if (tid < off) red[tid] += red[tid + off];
        __syncthreads();
    }
    const float rs = rsqrtf(red[0] / (float)DM + 1e-5f);
    for (int c = tid; c < DM; c += 256)
        xn[(size_t)row * DM + c] = (__bf16)(xr[c] * rs * w[c]);
}

// ------------------------------------------------------------------
// WMMA GEMM, register-blocked:
//   C[M,N](f32) = A[M,K](bf16 row-major) * B[N,K](bf16 row-major)^T
// One wave -> 64x32 macro-tile (4x2 grid of 16x16 WMMA tiles):
// per K-step loads 4 A-frags + 2 B-frags, issues 8 v_wmma (fragment reuse).
// Requires M % 64 == 0, N % 32 == 0, K % 32 == 0.
// Optional fused residual add (resid may be nullptr).
// ------------------------------------------------------------------
__global__ void wmma_gemm_bf16(const __bf16* __restrict__ A, int lda,
                               const __bf16* __restrict__ Bw, int ldb,
                               float* __restrict__ C, int ldc,
                               const float* __restrict__ resid,
                               int M, int N, int K) {
    const int wave = (blockIdx.x * blockDim.x + threadIdx.x) >> 5;
    const int lane = threadIdx.x & 31;
    const int nmt = N >> 5;                       // macro-tile columns
    const int tm = wave / nmt, tn = wave % nmt;
    if (tm >= (M >> 6)) return;                   // whole-wave guard: EXEC stays all-ones
    const int half = lane & 15, hi = lane >> 4;
    const int row0 = tm << 6, col0 = tn << 5;

    // A frag (16-bit 16x32 layout): lanes 0-15 hold K k0+0..7 / k0+16..23 of row `half`,
    // lanes 16-31 hold K k0+8..15 / k0+24..31.
    const __bf16* ap = A + (size_t)(row0 + half) * lda + hi * 8;
    // B frag: lane -> column; lanes 0-15 hold K k0+0..7 / k0+16..23, lanes 16-31 hold K k0+8..15 / k0+24..31.
    const __bf16* bp = Bw + (size_t)(col0 + half) * ldb + hi * 8;
    const size_t astr = (size_t)16 * lda;
    const size_t bstr = (size_t)16 * ldb;

    v8f acc[4][2] = {};
    for (int k = 0; k < K; k += 32) {
        v16bf a[4], b[2];
#pragma unroll
        for (int i = 0; i < 4; i++) {
            v8bf a0 = *(const v8bf*)(ap + i * astr + k);
            v8bf a1 = *(const v8bf*)(ap + i * astr + k + 16);
#pragma unroll
            for (int e = 0; e < 8; e++) { a[i][e] = a0[e]; a[i][e + 8] = a1[e]; }
        }
#pragma unroll
        for (int j = 0; j < 2; j++) {
            v8bf b0 = *(const v8bf*)(bp + j * bstr + k);
            v8bf b1 = *(const v8bf*)(bp + j * bstr + k + 16);
#pragma unroll
            for (int e = 0; e < 8; e++) { b[j][e] = b0[e]; b[j][e + 8] = b1[e]; }
        }
#pragma unroll
        for (int i = 0; i < 4; i++)
#pragma unroll
            for (int j = 0; j < 2; j++)
                acc[i][j] = __builtin_amdgcn_wmma_f32_16x16x32_bf16(
                    false, a[i], false, b[j], (short)0, acc[i][j], false, false);
    }
    // D layout: VGPR v -> row (+ 8*hi), col = half
#pragma unroll
    for (int i = 0; i < 4; i++)
#pragma unroll
        for (int j = 0; j < 2; j++)
#pragma unroll
            for (int v = 0; v < 8; v++) {
                size_t idx = (size_t)(row0 + i * 16 + hi * 8 + v) * ldc + col0 + j * 16 + half;
                C[idx] = acc[i][j][v] + (resid ? resid[idx] : 0.0f);
            }
}

// ------------------------------------------------------------------
// causal depthwise conv (d_conv=4, pad left 3) + bias + SiLU
// reads u = xr[:, 0:DI]; writes uc (f32) and ub (bf16)
// ------------------------------------------------------------------
__global__ void conv_silu_kernel(const float* __restrict__ xr, const float* __restrict__ cw,
                                 const float* __restrict__ cb,
                                 float* __restrict__ uc, __bf16* __restrict__ ub) {
    const int i = blockIdx.x * blockDim.x + threadIdx.x;   // over ROWS*DI
    if (i >= ROWS * DI) return;
    const int d = i & (DI - 1);
    const int row = i >> 10;                // b*LL + l
    const int l = row & (LL - 1);
    float acc = cb[d];
#pragma unroll
    for (int j = 0; j < 4; j++) {
        const int li = l - 3 + j;
        if (li >= 0)
            acc += cw[d * 4 + j] * xr[(size_t)(row - l + li) * (2 * DI) + d];
    }
    const float s = acc / (1.0f + __expf(-acc));   // SiLU
    uc[i] = s;
    ub[i] = (__bf16)s;
}

// ------------------------------------------------------------------
// delta = softplus(dt_raw + bias)
// ------------------------------------------------------------------
__global__ void softplus_kernel(float* __restrict__ delta, const float* __restrict__ bias, int n) {
    const int i = blockIdx.x * blockDim.x + threadIdx.x;
    if (i >= n) return;
    const float v = delta[i] + bias[i & (DI - 1)];
    delta[i] = (v > 20.0f) ? v : log1pf(__expf(v));
}

// ------------------------------------------------------------------
// selective scan: one lane per (b, d) channel, 16-state in VGPRs.
// grid = 8 blocks x 256 threads: block -> (b = bid>>2, d-range = (bid&3)*256)
// B_t / C_t (uniform per batch) broadcast via LDS each step — staged with the
// gfx1250 async global->LDS path when available.
// ------------------------------------------------------------------
__global__ void scan_kernel(const float* __restrict__ delta, const float* __restrict__ uc,
                            const float* __restrict__ xdbl, const float* __restrict__ A_log,
                            float* __restrict__ yscan) {
    __shared__ float sBC[2 * DS];
    const int b = blockIdx.x >> 2;
    const int d = ((blockIdx.x & 3) << 8) + threadIdx.x;
    float A[DS], h[DS];
#pragma unroll
    for (int n = 0; n < DS; n++) { A[n] = -__expf(A_log[d * DS + n]); h[n] = 0.f; }

    for (int t = 0; t < LL; t++) {
        const size_t r = (size_t)b * LL + t;
#ifdef HAVE_ASYNC_LDS
        if (threadIdx.x < 8) {     // 8 lanes x 16B = the 32 floats of (B_t, C_t)
            const float* g = xdbl + r * (DTR + 2 * DS) + DTR + threadIdx.x * 4;
            __builtin_amdgcn_global_load_async_to_lds_b128(
                (glob_v4i_p)(g), (lds_v4i_p)(&sBC[threadIdx.x * 4]), 0, 0);
        }
        __builtin_amdgcn_s_wait_asynccnt(0);
        __syncthreads();
#else
        if (threadIdx.x < 2 * DS) sBC[threadIdx.x] = xdbl[r * (DTR + 2 * DS) + DTR + threadIdx.x];
        __syncthreads();
#endif
        const float dt = delta[r * DI + d];
        const float uv = uc[r * DI + d];
        if (t + 1 < LL) {   // prefetch next step's per-channel operands (global_prefetch_b8)
            __builtin_prefetch(delta + (r + 1) * DI + d, 0, 0);
            __builtin_prefetch(uc + (r + 1) * DI + d, 0, 0);
        }
        const float du = dt * uv;
        float y = 0.f;
#pragma unroll
        for (int n = 0; n < DS; n++) {
            const float dA = __expf(dt * A[n]);
            h[n] = dA * h[n] + du * sBC[n];
            y += h[n] * sBC[DS + n];
        }
        yscan[r * DI + d] = y;
        __syncthreads();
    }
}

// ------------------------------------------------------------------
// y2 = (yscan + uc*D) * SiLU(res);  res = xr[:, DI:2*DI];  emit bf16 for out_proj
// ------------------------------------------------------------------
__global__ void gate_kernel(const float* __restrict__ yscan, const float* __restrict__ uc,
                            const float* __restrict__ Dw, const float* __restrict__ xr,
                            __bf16* __restrict__ y2b) {
    const int i = blockIdx.x * blockDim.x + threadIdx.x;
    if (i >= ROWS * DI) return;
    const int d = i & (DI - 1);
    const int row = i >> 10;
    const float res = xr[(size_t)row * (2 * DI) + DI + d];
    const float g = res / (1.0f + __expf(-res));
    y2b[i] = (__bf16)((yscan[i] + uc[i] * Dw[d]) * g);
}

// ------------------------------------------------------------------
extern "C" void kernel_launch(void* const* d_in, const int* in_sizes, int n_in,
                              void* d_out, int out_size, void* d_ws, size_t ws_size,
                              hipStream_t stream) {
    const float* x        = (const float*)d_in[0];   // (B,L,DM)
    const float* rms_w    = (const float*)d_in[1];   // (DM)
    const float* in_w     = (const float*)d_in[2];   // (2*DI, DM)
    const float* conv_w   = (const float*)d_in[3];   // (DI,1,4)
    const float* conv_b   = (const float*)d_in[4];   // (DI)
    const float* xproj_w  = (const float*)d_in[5];   // (DTR+2*DS, DI)
    const float* dt_w     = (const float*)d_in[6];   // (DI, DTR)
    const float* dt_b     = (const float*)d_in[7];   // (DI)
    const float* A_log    = (const float*)d_in[8];   // (DI, DS)
    const float* Dw       = (const float*)d_in[9];   // (DI)
    const float* out_w    = (const float*)d_in[10];  // (DM, DI)
    float* out            = (float*)d_out;           // (B,L,DM)

    // workspace carve-up (256B aligned)
    char*  ws = (char*)d_ws;
    size_t o  = 0;
    auto alloc = [&](size_t bytes) { o = (o + 255) & ~(size_t)255; size_t p = o; o += bytes; return p; };

    __bf16* xn_bf   = (__bf16*)(ws + alloc((size_t)ROWS * DM * 2));
    __bf16* w1_bf   = (__bf16*)(ws + alloc((size_t)2 * DI * DM * 2));
    float*  xr_f    = (float*) (ws + alloc((size_t)ROWS * 2 * DI * 4));
    float*  uc_f    = (float*) (ws + alloc((size_t)ROWS * DI * 4));
    __bf16* u_bf    = (__bf16*)(ws + alloc((size_t)ROWS * DI * 2));
    __bf16* xw_bf   = (__bf16*)(ws + alloc((size_t)(DTR + 2 * DS) * DI * 2));
    float*  xdbl_f  = (float*) (ws + alloc((size_t)ROWS * (DTR + 2 * DS) * 4));
    __bf16* xdbl_bf = (__bf16*)(ws + alloc((size_t)ROWS * (DTR + 2 * DS) * 2));
    __bf16* dtw_bf  = (__bf16*)(ws + alloc((size_t)DI * DTR * 2));
    float*  delta_f = (float*) (ws + alloc((size_t)ROWS * DI * 4));
    float*  yscan_f = (float*) (ws + alloc((size_t)ROWS * DI * 4));
    __bf16* y2_bf   = (__bf16*)(ws + alloc((size_t)ROWS * DI * 2));
    __bf16* ow_bf   = (__bf16*)(ws + alloc((size_t)DM * DI * 2));
    (void)ws_size; (void)in_sizes; (void)n_in; (void)out_size;

    auto cdiv = [](int a, int b) { return (a + b - 1) / b; };
    // waves = (M/64)*(N/32); 8 waves per 256-thread block
    auto gemm_blocks = [](int M, int N) { return (M / 64) * (N / 32) / 8; };

    // weight casts
    cast_f32_bf16<<<cdiv(2 * DI * DM, 256), 256, 0, stream>>>(in_w, w1_bf, 2 * DI * DM);
    cast_f32_bf16<<<cdiv((DTR + 2 * DS) * DI, 256), 256, 0, stream>>>(xproj_w, xw_bf, (DTR + 2 * DS) * DI);
    cast_f32_bf16<<<cdiv(DI * DTR, 256), 256, 0, stream>>>(dt_w, dtw_bf, DI * DTR);
    cast_f32_bf16<<<cdiv(DM * DI, 256), 256, 0, stream>>>(out_w, ow_bf, DM * DI);

    // 1) RMSNorm -> bf16
    rmsnorm_kernel<<<ROWS, 256, 0, stream>>>(x, rms_w, xn_bf);

    // 2) in_proj GEMM: xr[4096,2048] = xn @ in_w^T
    wmma_gemm_bf16<<<gemm_blocks(ROWS, 2 * DI), 256, 0, stream>>>(
        xn_bf, DM, w1_bf, DM, xr_f, 2 * DI, nullptr, ROWS, 2 * DI, DM);

    // 3) causal depthwise conv + SiLU
    conv_silu_kernel<<<cdiv(ROWS * DI, 256), 256, 0, stream>>>(xr_f, conv_w, conv_b, uc_f, u_bf);

    // 4) x_proj GEMM: x_dbl[4096,64] = u @ x_proj_w^T
    wmma_gemm_bf16<<<gemm_blocks(ROWS, DTR + 2 * DS), 256, 0, stream>>>(
        u_bf, DI, xw_bf, DI, xdbl_f, DTR + 2 * DS, nullptr, ROWS, DTR + 2 * DS, DI);

    // 5) dt_proj GEMM: delta_raw[4096,1024] = x_dbl[:, :32] @ dt_proj_w^T  (K=32 via lda=64)
    cast_f32_bf16<<<cdiv(ROWS * (DTR + 2 * DS), 256), 256, 0, stream>>>(
        xdbl_f, xdbl_bf, ROWS * (DTR + 2 * DS));
    wmma_gemm_bf16<<<gemm_blocks(ROWS, DI), 256, 0, stream>>>(
        xdbl_bf, DTR + 2 * DS, dtw_bf, DTR, delta_f, DI, nullptr, ROWS, DI, DTR);

    // 6) softplus(+bias)
    softplus_kernel<<<cdiv(ROWS * DI, 256), 256, 0, stream>>>(delta_f, dt_b, ROWS * DI);

    // 7) selective scan (one lane per channel, 16-state in VGPRs)
    scan_kernel<<<BB * (DI / 256), 256, 0, stream>>>(delta_f, uc_f, xdbl_f, A_log, yscan_f);

    // 8) skip + gate
    gate_kernel<<<cdiv(ROWS * DI, 256), 256, 0, stream>>>(yscan_f, uc_f, Dw, xr_f, y2_bf);

    // 9) out_proj GEMM with fused residual:  out = x + y2 @ out_w^T
    wmma_gemm_bf16<<<gemm_blocks(ROWS, DM), 256, 0, stream>>>(
        y2_bf, DI, ow_bf, DI, out, DM, x, ROWS, DM, DI);
}
